// DepthwiseMultiKernelAttention_25031069401677
// MI455X (gfx1250) — matrix operand order repaired and checked by reading, not compiled, not run on hardware
//
#include <hip/hip_runtime.h>
#include <stdint.h>

#define B_   8
#define C_   64
#define H_   256
#define W_   256
#define HW_  (H_ * W_)

typedef __attribute__((ext_vector_type(16))) __bf16 v16bf;
typedef __attribute__((ext_vector_type(8)))  float  v8f;

union ABf { uint32_t u[8]; v16bf v; };

struct F3 { float x, y, z; };   // 12B row segment -> global_load_b96-class load

__device__ __forceinline__ uint32_t f32_to_bf16_rne(float f) {
    uint32_t u = __float_as_uint(f);
    uint32_t r = u + 0x7FFFu + ((u >> 16) & 1u);
    return r >> 16;
}
__device__ __forceinline__ float bf16_to_f32(uint32_t h) {
    return __uint_as_float(h << 16);
}
// Split f32 pair into packed bf16 hi plane + bf16 residual (lo) plane.
__device__ __forceinline__ void split_pack(float d0, float d1,
                                           uint32_t& hi, uint32_t& lo) {
    uint32_t h0 = f32_to_bf16_rne(d0), h1 = f32_to_bf16_rne(d1);
    hi = h0 | (h1 << 16);
    uint32_t l0 = f32_to_bf16_rne(d0 - bf16_to_f32(h0));
    uint32_t l1 = f32_to_bf16_rne(d1 - bf16_to_f32(h1));
    lo = l0 | (l1 << 16);
}
__device__ __forceinline__ float conv9(const float* wg, F3 r0, F3 r1, F3 r2) {
    float d = wg[0] * r0.x;
    d = fmaf(wg[1], r0.y, d);
    d = fmaf(wg[2], r0.z, d);
    d = fmaf(wg[3], r1.x, d);
    d = fmaf(wg[4], r1.y, d);
    d = fmaf(wg[5], r1.z, d);
    d = fmaf(wg[6], r2.x, d);
    d = fmaf(wg[7], r2.y, d);
    d = fmaf(wg[8], r2.z, d);
    return d;
}

// ---------------------------------------------------------------------------
// Stage 1 (1 workgroup): M = softmax(G) @ P @ softmax(A), then write M to
// scratch pre-swizzled into WMMA A-operand layout (16x32 bf16 tiles,
// ISA 7.12.2), as separate hi (dwords [0,2048)) and lo ([2048,4096)) planes.
// ---------------------------------------------------------------------------
__global__ __launch_bounds__(256)
void prep_mix_kernel(const float* __restrict__ attW,
                     const float* __restrict__ gloW,
                     const float* __restrict__ pntW,
                     uint32_t*    __restrict__ wsA) {
    __shared__ float As[C_ * C_];
    __shared__ float Gs[C_ * C_];
    __shared__ float Ts[C_ * C_];
    const int t = threadIdx.x;

    if (t < 128) {   // row softmaxes: 0..63 -> A', 64..127 -> G'
        const float* src = (t < 64) ? (attW + t * C_) : (gloW + (t - 64) * C_);
        float*       dst = (t < 64) ? (As + t * C_)   : (Gs + (t - 64) * C_);
        float mx = -3.4e38f;
        for (int j = 0; j < C_; ++j) mx = fmaxf(mx, src[j]);
        float s = 0.f;
        for (int j = 0; j < C_; ++j) { float e = __expf(src[j] - mx); dst[j] = e; s += e; }
        float inv = 1.f / s;
        for (int j = 0; j < C_; ++j) dst[j] *= inv;
    }
    __syncthreads();

    for (int e = 0; e < 16; ++e) {          // T = P @ A'
        int idx = t * 16 + e;
        int i = idx >> 6, j = idx & 63;
        float acc = 0.f;
        for (int c = 0; c < C_; ++c) acc += pntW[i * C_ + c] * As[c * C_ + j];
        Ts[idx] = acc;
    }
    __syncthreads();

    for (int e = 0; e < 16; ++e) {          // M = G' @ T  (overwrites As)
        int idx = t * 16 + e;
        int i = idx >> 6, j = idx & 63;
        float acc = 0.f;
        for (int c = 0; c < C_; ++c) acc += Gs[i * C_ + c] * Ts[c * C_ + j];
        As[idx] = acc;
    }
    __syncthreads();

    // A-operand swizzle: tile = m*2+kt; dword in tile = lane*8 + j.
    // K(lane,j,half) = (j&4 ? 16:0) + (lane&16 ? 8:0) + (j&3)*2 + half.
    for (int i = t; i < 2048; i += 256) {
        int tile = i >> 8, rem = i & 255, lane = rem >> 3, j = rem & 7;
        int m = tile >> 1, kt = tile & 1;
        int row = m * 16 + (lane & 15);
        int kb  = ((j & 4) ? 16 : 0) + ((lane & 16) ? 8 : 0) + (j & 3) * 2;
        int c0  = kt * 32 + kb;
        split_pack(As[row * 64 + c0], As[row * 64 + c0 + 1], wsA[i], wsA[2048 + i]);
    }
}

// ---------------------------------------------------------------------------
// Stage 2: fused depthwise 3x3 conv + 64x64 channel mix via split-bf16 WMMA.
// Block = 256 threads = 8 waves; wave owns 16 pixels of one row, all 64 ch.
// Interior waves (~87%) take a mask-free path with 12B row-vector loads;
// border waves use clamped-address + {0,1}-mask scalar taps.
// ---------------------------------------------------------------------------
__global__ __launch_bounds__(256)
void fused_dwconv_mix_kernel(const float*    __restrict__ x,
                             const float*    __restrict__ dwW,
                             const uint32_t* __restrict__ wsA,
                             float*          __restrict__ out) {
    __shared__ uint32_t Mlds[4096];   // 16 KB: pre-swizzled M (hi+lo planes)
    __shared__ float    Wlds[C_ * 9]; // depthwise weights

    const int tid = threadIdx.x;
    for (int i = tid; i < 4096;   i += 256) Mlds[i] = wsA[i];
    for (int i = tid; i < C_ * 9; i += 256) Wlds[i] = dwW[i];
    __syncthreads();

    const int blk  = blockIdx.x;         // grid = B * H * 2
    const int b    = blk >> 9;
    const int rem  = blk & 511;
    const int h    = rem >> 1;
    const int wave = tid >> 5, lane = tid & 31;
    const int w0   = (rem & 1) * 128 + wave * 16;
    const int p    = lane & 15;
    const int w    = w0 + p;

    const float* xb = x + (size_t)b * C_ * HW_;

    // B operands (32x16 bf16, K x N): N=lane&15, K=(j&4?16:0)+(lane&16?8:0)+(j&3)*2 (+1);
    // matches A-operand table order.  Channel c = kt*32 + K.
    ABf Bhi[2], Blo[2];

    const bool interior = (h > 0) && (h < H_ - 1) && (w0 > 0) && (w0 + 16 < W_);
    if (interior) {
        // Mask-free fast path: 3 contiguous taps per row -> one 12B load.
        const float* xrow = xb + (size_t)(h - 1) * W_ + (w - 1);
        #pragma unroll
        for (int kt = 0; kt < 2; ++kt) {
            #pragma unroll
            for (int j = 0; j < 8; ++j) {
                int cbase = kt * 32 + ((j & 4) ? 16 : 0) + ((lane & 16) ? 8 : 0) + ((j & 3) << 1);
                const float* p0  = xrow + (size_t)cbase * HW_;
                const float* p1  = p0 + HW_;
                const float* wg0 = Wlds + cbase * 9;
                F3 a0 = *(const F3*)(p0);
                F3 a1 = *(const F3*)(p0 + W_);
                F3 a2 = *(const F3*)(p0 + 2 * W_);
                F3 b0 = *(const F3*)(p1);
                F3 b1 = *(const F3*)(p1 + W_);
                F3 b2 = *(const F3*)(p1 + 2 * W_);
                float d0 = conv9(wg0,     a0, a1, a2);
                float d1 = conv9(wg0 + 9, b0, b1, b2);
                split_pack(d0, d1, Bhi[kt].u[j], Blo[kt].u[j]);
            }
        }
    } else {
        // Border path: clamped address + {0,1} mask (branch-free per tap).
        float msk[9]; int off[9];
        #pragma unroll
        for (int dh = -1; dh <= 1; ++dh) {
            #pragma unroll
            for (int dc = -1; dc <= 1; ++dc) {
                int k  = (dh + 1) * 3 + (dc + 1);
                int hh = h + dh, ww = w + dc;
                bool ok = (hh >= 0) && (hh < H_) && (ww >= 0) && (ww < W_);
                int hc = min(max(hh, 0), H_ - 1);
                int wc = min(max(ww, 0), W_ - 1);
                msk[k] = ok ? 1.f : 0.f;
                off[k] = hc * W_ + wc;
            }
        }
        #pragma unroll
        for (int kt = 0; kt < 2; ++kt) {
            #pragma unroll
            for (int j = 0; j < 8; ++j) {
                int cbase = kt * 32 + ((j & 4) ? 16 : 0) + ((lane & 16) ? 8 : 0) + ((j & 3) << 1);
                const float* x0  = xb + (size_t)cbase * HW_;
                const float* x1  = x0 + HW_;
                const float* wg0 = Wlds + cbase * 9;
                float d0 = 0.f, d1 = 0.f;
                #pragma unroll
                for (int k = 0; k < 9; ++k) {
                    d0 = fmaf(wg0[k]     * msk[k], x0[off[k]], d0);
                    d1 = fmaf(wg0[9 + k] * msk[k], x1[off[k]], d1);
                }
                split_pack(d0, d1, Bhi[kt].u[j], Blo[kt].u[j]);
            }
        }
    }

    // Channel-mix GEMM: per out-tile, 2 K-tiles x 3 split-bf16 products.
    const size_t outbase = (size_t)b * C_ * HW_ + (size_t)h * W_ + w;
    #pragma unroll
    for (int m = 0; m < 4; ++m) {
        v8f acc = {};
        #pragma unroll
        for (int kt = 0; kt < 2; ++kt) {
            ABf Ah, Al;
            int base = ((m * 2 + kt) * 32 + lane) * 8;
            #pragma unroll
            for (int j = 0; j < 8; ++j) {
                Ah.u[j] = Mlds[base + j];
                Al.u[j] = Mlds[2048 + base + j];
            }
            acc = __builtin_amdgcn_wmma_f32_16x16x32_bf16(
                      false, Ah.v, false, Blo[kt].v, (short)0, acc, false, false);
            acc = __builtin_amdgcn_wmma_f32_16x16x32_bf16(
                      false, Al.v, false, Bhi[kt].v, (short)0, acc, false, false);
            acc = __builtin_amdgcn_wmma_f32_16x16x32_bf16(
                      false, Ah.v, false, Bhi[kt].v, (short)0, acc, false, false);
        }
        // D layout: VGPR j -> out-ch m*16 + j (+8 for lanes 16-31), N = lane&15.
        #pragma unroll
        for (int j = 0; j < 8; ++j) {
            int c = m * 16 + j + ((lane & 16) ? 8 : 0);
            out[outbase + (size_t)c * HW_] = acc[j];
        }
    }
}

extern "C" void kernel_launch(void* const* d_in, const int* in_sizes, int n_in,
                              void* d_out, int out_size, void* d_ws, size_t ws_size,
                              hipStream_t stream) {
    (void)in_sizes; (void)n_in; (void)out_size; (void)ws_size;
    const float* x    = (const float*)d_in[0]; // [8,64,256,256]
    const float* dwW  = (const float*)d_in[1]; // [64,1,3,3]
    const float* pntW = (const float*)d_in[2]; // [64,64,1,1]
    const float* attW = (const float*)d_in[3]; // [64,64]
    const float* gloW = (const float*)d_in[4]; // [64,64]
    uint32_t* wsA = (uint32_t*)d_ws;           // 16 KB used
    float*    out = (float*)d_out;

    prep_mix_kernel<<<1, 256, 0, stream>>>(attW, gloW, pntW, wsA);
    fused_dwconv_mix_kernel<<<B_ * H_ * 2, 256, 0, stream>>>(x, dwW, wsA, out);
}
